// EdgeUpdate_9122510536849
// MI455X (gfx1250) — matrix operand order repaired and checked by reading, not compiled, not run on hardware
//
#include <hip/hip_runtime.h>
#include <hip/hip_bf16.h>
#include <math.h>

typedef _Float16 f16;
typedef __attribute__((ext_vector_type(16))) _Float16 v16h;
typedef __attribute__((ext_vector_type(8)))  float    v8f;

#define NB_  63      // neighbors per atom
#define MPAD 64      // padded row tile
#define NAB  128
#define NSB  128
#define DIN  384     // 2*NAB + NSB
#define DHID 256     // NAB + NSB
#define DOUT 128

// shifted softplus: log(1+exp(x)) - log(2), numerically stable
__device__ __forceinline__ float ssp(float v) {
    float ax = fabsf(v);
    return fmaxf(v, 0.0f) + log1pf(expf(-ax)) - 0.69314718055994530942f;
}

// load two consecutive f16 (one 32-bit word) into fragment slots 2r, 2r+1
__device__ __forceinline__ void ld_pair(v16h& frag, int r, const f16* p) {
    union { unsigned u; f16 h[2]; } t;
    t.u = *(const unsigned*)p;
    frag[2*r]   = t.h[0];
    frag[2*r+1] = t.h[1];
}

// Pre-pass: convert W1 (384x256 f32, row-major) -> W1t (256x384 f16, N-major)
//           convert W2 (256x128 f32, row-major) -> W2t (128x256 f16, N-major)
__global__ __launch_bounds__(256) void convert_weights_kernel(
    const float* __restrict__ W1, const float* __restrict__ W2,
    f16* __restrict__ W1t, f16* __restrict__ W2t)
{
    int i = blockIdx.x * blockDim.x + threadIdx.x;
    if (i < DIN * DHID) {
        int d = i / DHID, h = i - d * DHID;
        W1t[h * DIN + d] = (f16)W1[i];
    }
    if (i < DHID * DOUT) {
        int d = i / DOUT, h = i - d * DOUT;
        W2t[h * DHID + d] = (f16)W2[i];
    }
}

__global__ __launch_bounds__(256) void edge_update_kernel(
    const float* __restrict__ x,      // (64,64,128)
    const int*   __restrict__ nbr,    // (64,64,63)
    const float* __restrict__ f_ij,   // (64,64,63,128)
    const float* __restrict__ b1,     // (256)
    const float* __restrict__ b2,     // (128)
    const f16*   __restrict__ W1t,    // (256,384) n-major f16
    const f16*   __restrict__ W2t,    // (128,256) n-major f16
    float*       __restrict__ out)    // (64,64,63,128)
{
    extern __shared__ char smem[];
    f16* Ash = (f16*)smem;                                // [64][384] f16 = 48 KB
    f16* H1  = (f16*)(smem + MPAD * DIN * 2);             // [64][256] f16 = 32 KB
    int* nbs = (int*)(smem + (MPAD * DIN + MPAD * DHID) * 2);

    const int tid  = threadIdx.x;
    const int bidx = blockIdx.x;      // b*64 + a
    const int b    = bidx >> 6;

    // keep the L2-resident weight images warm
    __builtin_prefetch(&W1t[(size_t)tid * DIN], 0, 1);
    __builtin_prefetch(&W2t[(size_t)(tid & 127) * DHID], 0, 1);

    if (tid < MPAD) nbs[tid] = (tid < NB_) ? nbr[(size_t)bidx * NB_ + tid] : 0;
    __syncthreads();

    // ---- stage A = [f_ij | x[neighbor] | x[self]] into LDS as f16 ----
    const float* xi  = x    + (size_t)bidx * NAB;
    const float* fij = f_ij + (size_t)bidx * NB_ * NSB;
    for (int e = tid; e < MPAD * DIN; e += 256) {
        int row = e / DIN;
        int col = e - row * DIN;
        float v = 0.0f;
        if (row < NB_) {
            if (col < NSB)            v = fij[row * NSB + col];
            else if (col < NSB + NAB) v = x[(((size_t)b << 6) + nbs[row]) * NAB + (col - NSB)];
            else                      v = xi[col - (NSB + NAB)];
        }
        Ash[e] = (f16)v;
    }
    __syncthreads();

    const int lane = tid & 31;
    const int wid  = tid >> 5;        // 0..7
    const int lh   = lane >> 4;       // lane half (0/1)
    const int ln   = lane & 15;
    const int mt   = (wid & 3) << 4;  // row tile base: 0/16/32/48
    const int ng   = wid >> 2;        // N-group (0/1)

    // ---- GEMM1: (64x384) x (384x256), each wave: 1 M-tile x 8 N-tiles ----
    v8f acc[8];
    #pragma unroll
    for (int j = 0; j < 8; ++j)
        acc[j] = (v8f){0.f, 0.f, 0.f, 0.f, 0.f, 0.f, 0.f, 0.f};

    for (int kb = 0; kb < DIN; kb += 32) {
        v16h afrag;
        #pragma unroll
        for (int r = 0; r < 8; ++r) {
            // A 16x32 f16 layout: K0 = (r<4?0:16) + 8*laneHalf + 2*(r%4)
            int k0 = ((r & 4) << 2) + (lh << 3) + ((r & 3) << 1);
            ld_pair(afrag, r, &Ash[(mt + ln) * DIN + kb + k0]);
        }
        #pragma unroll
        for (int j = 0; j < 8; ++j) {
            const int ncol = ng * 128 + j * 16 + ln;
            v16h bfrag;
            #pragma unroll
            for (int r = 0; r < 8; ++r) {
                // B 32x16 layout: K0 = (r<4?0:16) + 8*laneHalf + 2*(r%4)
                int k0 = ((r & 4) << 2) + (lh << 3) + ((r & 3) << 1);
                ld_pair(bfrag, r, &W1t[(size_t)ncol * DIN + kb + k0]);
            }
            acc[j] = __builtin_amdgcn_wmma_f32_16x16x32_f16(
                false, afrag, false, bfrag, (short)0, acc[j], false, false);
        }
    }

    // ---- epilogue 1: bias + shifted softplus, write H1 to LDS as f16 ----
    #pragma unroll
    for (int j = 0; j < 8; ++j) {
        const int col  = ng * 128 + j * 16 + ln;
        const float bv = b1[col];
        #pragma unroll
        for (int r = 0; r < 8; ++r) {
            int row = mt + r + (lh << 3);   // C/D layout: M = r + 8*laneHalf
            H1[row * DHID + col] = (f16)ssp(acc[j][r] + bv);
        }
    }
    __syncthreads();

    // ---- GEMM2: (64x256) x (256x128), each wave: 1 M-tile x 4 N-tiles ----
    v8f acc2[4];
    #pragma unroll
    for (int j = 0; j < 4; ++j)
        acc2[j] = (v8f){0.f, 0.f, 0.f, 0.f, 0.f, 0.f, 0.f, 0.f};

    for (int kb = 0; kb < DHID; kb += 32) {
        v16h afrag;
        #pragma unroll
        for (int r = 0; r < 8; ++r) {
            int k0 = ((r & 4) << 2) + (lh << 3) + ((r & 3) << 1);
            ld_pair(afrag, r, &H1[(mt + ln) * DHID + kb + k0]);
        }
        #pragma unroll
        for (int j = 0; j < 4; ++j) {
            const int ncol = ng * 64 + j * 16 + ln;
            v16h bfrag;
            #pragma unroll
            for (int r = 0; r < 8; ++r) {
                int k0 = ((r & 4) << 2) + (lh << 3) + ((r & 3) << 1);
                ld_pair(bfrag, r, &W2t[(size_t)ncol * DHID + kb + k0]);
            }
            acc2[j] = __builtin_amdgcn_wmma_f32_16x16x32_f16(
                false, afrag, false, bfrag, (short)0, acc2[j], false, false);
        }
    }

    // ---- epilogue 2: bias + shifted softplus, store f32 output ----
    #pragma unroll
    for (int j = 0; j < 4; ++j) {
        const int col  = ng * 64 + j * 16 + ln;
        const float bv = b2[col];
        #pragma unroll
        for (int r = 0; r < 8; ++r) {
            int row = mt + r + (lh << 3);
            if (row < NB_)
                out[((size_t)bidx * NB_ + row) * DOUT + col] = ssp(acc2[j][r] + bv);
        }
    }
}

extern "C" void kernel_launch(void* const* d_in, const int* in_sizes, int n_in,
                              void* d_out, int out_size, void* d_ws, size_t ws_size,
                              hipStream_t stream) {
    const float* x    = (const float*)d_in[0];
    const int*   nbr  = (const int*)  d_in[1];
    const float* f_ij = (const float*)d_in[2];
    const float* W1   = (const float*)d_in[3];
    const float* b1   = (const float*)d_in[4];
    const float* W2   = (const float*)d_in[5];
    const float* b2   = (const float*)d_in[6];
    float* out = (float*)d_out;

    f16* W1t = (f16*)d_ws;                                   // 256x384 f16 = 192 KB
    f16* W2t = (f16*)((char*)d_ws + (size_t)DIN * DHID * 2); // 128x256 f16 =  64 KB

    // weight convert+transpose (L2-resident, reused by all 4096 blocks)
    convert_weights_kernel<<<(DIN * DHID + 255) / 256, 256, 0, stream>>>(W1, W2, W1t, W2t);

    const size_t lds_bytes = (size_t)(MPAD * DIN + MPAD * DHID) * 2 + MPAD * sizeof(int);
    edge_update_kernel<<<64 * 64, 256, lds_bytes, stream>>>(
        x, nbr, f_ij, b1, b2, W1t, W2t, out);
}
